// ClusteringModule_68693706932683
// MI455X (gfx1250) — matrix operand order repaired and checked by reading, not compiled, not run on hardware
//
#include <hip/hip_runtime.h>

typedef _Float16 v16h __attribute__((ext_vector_type(16)));
typedef _Float16 v8h  __attribute__((ext_vector_type(8)));
typedef float    v8f  __attribute__((ext_vector_type(8)));
typedef float    v4f  __attribute__((ext_vector_type(4)));
typedef unsigned int u32x4 __attribute__((ext_vector_type(4)));
typedef unsigned int u32x8 __attribute__((ext_vector_type(8)));

#define NCLUS 128
#define DIM   256
#define LDS_ROW 264          // 256 f16 data + 8 f16 (4 DWORD) TDM pad per row
#define LDS_BYTES (NCLUS * LDS_ROW * 2)   // 67584
#define ROWS_PER_BLOCK 128   // 8 waves x 16 rows

__device__ __forceinline__ float sq4(v4f v) {
    return v.x * v.x + v.y * v.y + v.z * v.z + v.w * v.w;
}

// Kernel 1: per-centroid squared norms (f32) + f16 row-major copy of centroids.
__global__ void prep_kernel(const float* __restrict__ cent,
                            float* __restrict__ csq,
                            _Float16* __restrict__ centH) {
    const int tid = threadIdx.x;   // 256 threads, 1 block
    #pragma unroll 4
    for (int i = 0; i < (NCLUS * DIM) / 256; ++i) {
        int idx = tid + i * 256;
        centH[idx] = (_Float16)cent[idx];
    }
    if (tid < NCLUS) {
        const v4f* row = (const v4f*)(cent + (size_t)tid * DIM);
        float s = 0.f;
        #pragma unroll 4
        for (int i = 0; i < DIM / 4; ++i) s += sq4(row[i]);
        csq[tid] = s;
    }
}

// Kernel 2: TDM-staged centroids + WMMA GEMM + student-t epilogue + argmax.
__launch_bounds__(256)
__global__ void cluster_kernel(const float* __restrict__ z,
                               const _Float16* __restrict__ centH,
                               const float* __restrict__ csq_g,
                               float* __restrict__ out_s,
                               float* __restrict__ out_c) {
    extern __shared__ _Float16 cenL[];   // dynamic LDS: 128 rows x 264 f16 (padded)

    const int tid  = threadIdx.x;
    const int wave = tid >> 5;
    const int lane = tid & 31;
    const int half = lane >> 4;
    const int l16  = lane & 15;

    // ---- Tensor Data Mover: DMA 128x256 f16 centroid tile into LDS, with
    //      4-DWORD padding after every 128-DWORD row (bank-conflict-free reads).
    if (tid < 32) {   // wave 0 issues the DMA
        unsigned long long ga = (unsigned long long)(uintptr_t)centH;
        u32x4 g0;
        g0.x = 1u;                                  // count=1 valid descriptor
        g0.y = 0u;                                  // lds_addr = 0 (dynamic LDS base)
        g0.z = (unsigned)(ga & 0xffffffffu);        // global_addr[31:0]
        g0.w = (unsigned)(ga >> 32) | (2u << 30);   // global_addr[56:32] | type=2
        u32x8 g1;
        g1[0] = (1u << 16)          // data_size = 2 bytes
              | (1u << 20)          // pad_enable
              | (6u << 22)          // pad_interval: 128 DWORDs (one 512B row)
              | (3u << 25);         // pad_amount: 4 DWORDs (16B)
        g1[1] = ((unsigned)DIM & 0xffffu) << 16;    // tensor_dim0 = 256 (lo16)
        g1[2] = ((unsigned)NCLUS & 0xffffu) << 16;  // dim0 hi=0 | tensor_dim1 = 128 (lo16)
        g1[3] = ((unsigned)DIM) << 16;              // dim1 hi=0 | tile_dim0 = 256
        g1[4] = (unsigned)NCLUS;                    // tile_dim1 = 128, tile_dim2 = 0
        g1[5] = (unsigned)DIM;                      // tensor_dim0_stride = 256
        g1[6] = 0u;                                 // stride0 hi | dim1_stride lo
        g1[7] = 0u;
        u32x4 gz = {0u, 0u, 0u, 0u};                // groups 2/3: zeros (2-D tensor)
        asm volatile("tensor_load_to_lds %0, %1, %2, %3"
                     :: "s"(g0), "s"(g1), "s"(gz), "s"(gz)
                     : "memory");
        __builtin_amdgcn_s_wait_tensorcnt(0);
    }
    __syncthreads();

    const size_t rowBase = (size_t)blockIdx.x * ROWS_PER_BLOCK + (size_t)wave * 16;
    const float* zrow = z + (rowBase + (size_t)l16) * DIM;

    v8f acc[8];
    #pragma unroll
    for (int nt = 0; nt < 8; ++nt)
        #pragma unroll
        for (int r = 0; r < 8; ++r) acc[nt][r] = 0.f;

    float zsqPart = 0.f;

    // ---- K loop: 8 steps of 32 (unroll 2: bounded register pipeline) ----
    #pragma unroll 2
    for (int ks = 0; ks < 8; ++ks) {
        const int k0 = ks * 32;
        v4f a0 = __builtin_nontemporal_load((const v4f*)(zrow + k0 + half * 8));
        v4f a1 = __builtin_nontemporal_load((const v4f*)(zrow + k0 + half * 8 + 4));
        v4f a2 = __builtin_nontemporal_load((const v4f*)(zrow + k0 + 16 + half * 8));
        v4f a3 = __builtin_nontemporal_load((const v4f*)(zrow + k0 + 16 + half * 8 + 4));
        zsqPart += sq4(a0) + sq4(a1) + sq4(a2) + sq4(a3);

        v16h a;
        a[0]  = (_Float16)a0.x; a[1]  = (_Float16)a0.y; a[2]  = (_Float16)a0.z; a[3]  = (_Float16)a0.w;
        a[4]  = (_Float16)a1.x; a[5]  = (_Float16)a1.y; a[6]  = (_Float16)a1.z; a[7]  = (_Float16)a1.w;
        a[8]  = (_Float16)a2.x; a[9]  = (_Float16)a2.y; a[10] = (_Float16)a2.z; a[11] = (_Float16)a2.w;
        a[12] = (_Float16)a3.x; a[13] = (_Float16)a3.y; a[14] = (_Float16)a3.z; a[15] = (_Float16)a3.w;

        #pragma unroll
        for (int nt = 0; nt < 8; ++nt) {
            const int crow = nt * 16 + l16;        // centroid row = output column
            const _Float16* bp = cenL + crow * LDS_ROW + k0 + half * 8;
            v8h b0 = *(const v8h*)bp;              // K = k0+half*8 .. +7
            v8h b1 = *(const v8h*)(bp + 16);       // K = k0+16+half*8 .. +7
            v16h b;
            #pragma unroll
            for (int i = 0; i < 8; ++i) { b[i] = b0[i]; b[8 + i] = b1[i]; }
            acc[nt] = __builtin_amdgcn_wmma_f32_16x16x32_f16(
                false, a, false, b, (short)0, acc[nt], false, false);
        }
    }

    // ---- ||z||^2: combine K-halves, gather the 8 rows this lane owns ----
    zsqPart += __shfl_xor(zsqPart, 16, 32);
    float zq[8];
    #pragma unroll
    for (int r = 0; r < 8; ++r) zq[r] = __shfl(zsqPart, (half << 3) + r, 32);

    // ---- Student-t epilogue (alpha=1): st = 1/(1 + ||z-c||) ----
    float psum[8], bestV[8];
    int bestI[8];
    #pragma unroll
    for (int r = 0; r < 8; ++r) { psum[r] = 0.f; bestV[r] = -1e30f; bestI[r] = 0; }

    #pragma unroll
    for (int nt = 0; nt < 8; ++nt) {
        const int n = nt * 16 + l16;
        const float cq = csq_g[n];
        #pragma unroll
        for (int r = 0; r < 8; ++r) {
            float sqd = fmaxf(zq[r] + cq - 2.f * acc[nt][r], 0.f);
            float st  = 1.f / (1.f + sqrtf(sqd));
            acc[nt][r] = st;
            psum[r] += st;
            if (st > bestV[r]) { bestV[r] = st; bestI[r] = n; }
        }
    }

    // ---- Half-wave butterfly reductions (M halves independent) ----
    #pragma unroll
    for (int m = 1; m < 16; m <<= 1) {
        #pragma unroll
        for (int r = 0; r < 8; ++r) {
            psum[r] += __shfl_xor(psum[r], m, 32);
            float ov = __shfl_xor(bestV[r], m, 32);
            int   oi = __shfl_xor(bestI[r], m, 32);
            if (ov > bestV[r] || (ov == bestV[r] && oi < bestI[r])) {
                bestV[r] = ov; bestI[r] = oi;
            }
        }
    }

    float rinv[8];
    #pragma unroll
    for (int r = 0; r < 8; ++r) rinv[r] = 1.f / psum[r];

    // ---- Stores (non-temporal: streamed once) ----
    #pragma unroll
    for (int nt = 0; nt < 8; ++nt) {
        #pragma unroll
        for (int r = 0; r < 8; ++r) {
            size_t row = rowBase + (size_t)(half * 8 + r);
            __builtin_nontemporal_store(acc[nt][r] * rinv[r],
                                        out_s + row * NCLUS + nt * 16 + l16);
        }
    }
    if (l16 == 0) {
        #pragma unroll
        for (int r = 0; r < 8; ++r) {
            size_t row = rowBase + (size_t)(half * 8 + r);
            out_c[row] = (float)bestI[r];
        }
    }
}

extern "C" void kernel_launch(void* const* d_in, const int* in_sizes, int n_in,
                              void* d_out, int out_size, void* d_ws, size_t ws_size,
                              hipStream_t stream) {
    const float* z    = (const float*)d_in[0];
    const float* cent = (const float*)d_in[1];
    float* out        = (float*)d_out;

    float*    csq   = (float*)d_ws;                          // 128 f32
    _Float16* centH = (_Float16*)((char*)d_ws + 512);        // 128x256 f16 (64 KB)

    const int B = in_sizes[0] / DIM;          // 131072 rows
    float* out_s = out;
    float* out_c = out + (size_t)B * NCLUS;   // concatenated tuple output

    prep_kernel<<<1, 256, 0, stream>>>(cent, csq, centH);
    cluster_kernel<<<B / ROWS_PER_BLOCK, 256, LDS_BYTES, stream>>>(
        z, centH, csq, out_s, out_c);
}
